// MultiHeadAttention_72447508348954
// MI455X (gfx1250) — compile-verified, matrix operand order checked by reading, not run on hardware
//
#include <hip/hip_runtime.h>
#include <hip/hip_bf16.h>
#include <stdint.h>

// ---------------------------------------------------------------------------
// MI455X (gfx1250) multi-head attention forward.
// wave32; v_wmma_f32_16x16x32_bf16 for every GEMM (f32 accumulate);
// global_load_async_to_lds_b128 staging for the bf16 attention tiles;
// global_prefetch for the projection GEMM K-stream.
// d_out = [ out (B*L*D f32) | attn (B*H*L*L f32) ]
// ---------------------------------------------------------------------------

typedef __attribute__((ext_vector_type(16))) __bf16 v16bf;
typedef __attribute__((ext_vector_type(8)))  float  v8f;

constexpr int Bc = 4, Lc = 2048, Dc = 1024, Hc = 16, DKc = 64, DVc = 64;

// LDS tile strides (in ushorts). Chosen so every fragment chunk is 16B-aligned
// (stride*2 % 16 == 0) and 16 consecutive rows hit distinct banks
// (20*r mod 64 and 36*r mod 64 are permutations for r=0..15).
constexpr int KS32 = 40;   // tiles with 32 K-columns  (80 B rows)
constexpr int KS64 = 72;   // tiles with 64 K-columns  (144 B rows)

__device__ __forceinline__ unsigned short f32_to_bf16(float f) {
    unsigned int u = __float_as_uint(f);
    unsigned int r = u + 0x7FFFu + ((u >> 16) & 1u);   // round-to-nearest-even
    return (unsigned short)(r >> 16);
}

__device__ __forceinline__ unsigned pack_bf16x2(float lo, float hi) {
    return (unsigned)f32_to_bf16(lo) | ((unsigned)f32_to_bf16(hi) << 16);
}

// Low 32 bits of a generic pointer to __shared__ data = LDS byte offset.
__device__ __forceinline__ unsigned lds_offset(const void* p) {
    return (unsigned)(uintptr_t)p;
}

// Async byte copy global -> LDS, 16 bytes per lane (ASYNCcnt-tracked).
__device__ __forceinline__ void async_copy_b128(unsigned lds_byte_off,
                                                const void* gaddr) {
    asm volatile("global_load_async_to_lds_b128 %0, %1, off"
                 :: "v"(lds_byte_off), "v"(gaddr)
                 : "memory");
}

__device__ __forceinline__ void wait_async_all() {
    asm volatile("s_wait_asynccnt 0x0" ::: "memory");
}

// 16x32 bf16 operand fragment from an LDS tile stored row-major [row][k]
// (rows = M for the A operand, rows = N for the B operand, i.e. B^T).
// Per ISA 7.12.2: lane<16 -> row=lane,   K = {0..7, 16..23} (+kcol_base)
//                 lane>=16 -> row=lane-16, K = {8..15, 24..31}
// Both 8-element K-groups are 16B contiguous -> two ds_load_b128.
__device__ __forceinline__ v16bf load_frag(const unsigned short* smem,
                                           int row_base, int kcol_base,
                                           int stride, int lane) {
    int row = row_base + (lane & 15);
    int kb  = kcol_base + ((lane >> 4) << 3);
    const unsigned short* p = smem + row * stride + kb;
    union { v16bf v; uint4 q[2]; } f;
    f.q[0] = *(const uint4*)(p);
    f.q[1] = *(const uint4*)(p + 16);
    return f.v;
}

__device__ __forceinline__ v8f wmma_bf16(v16bf a, v16bf b, v8f c) {
    return __builtin_amdgcn_wmma_f32_16x16x32_bf16(
        /*neg_a=*/false, a, /*neg_b=*/false, b,
        /*c_mod=*/(short)0, c, /*reuse_a=*/false, /*reuse_b=*/false);
}

// ---------------------------------------------------------------------------
// Generic GEMM: out[M,N] = (A[M,K] @ W[K,N] + bias[N]) * scale  (+ residual)
// 256 threads = 8 waves; block tile 128x128; wave tile 32x64; K-step 32.
// ---------------------------------------------------------------------------
template <bool BF16_OUT, bool ADD_RESIDUAL>
__global__ __launch_bounds__(256) void gemm_kernel(
    const float* __restrict__ A, const float* __restrict__ W,
    const float* __restrict__ bias, const float* __restrict__ residual,
    void* __restrict__ outp, int M, int N, int K, float scale) {
    __shared__ unsigned short As[128 * KS32];   // [m][k]
    __shared__ unsigned short Bs[128 * KS32];   // [n][k] = W^T tile

    const int tid  = threadIdx.x;
    const int lane = tid & 31;
    const int wave = tid >> 5;
    const int m0 = blockIdx.y * 128;
    const int n0 = blockIdx.x * 128;
    const int wm = (wave >> 1) * 32;
    const int wn = (wave & 1) * 64;

    v8f acc[2][4];
#pragma unroll
    for (int mi = 0; mi < 2; ++mi)
#pragma unroll
        for (int ni = 0; ni < 4; ++ni)
            acc[mi][ni] = (v8f){0.f, 0.f, 0.f, 0.f, 0.f, 0.f, 0.f, 0.f};

    for (int k0 = 0; k0 < K; k0 += 32) {
        // Stage A 128x32 (f32 -> bf16) with float4 loads / uint2 LDS stores
        for (int i = tid; i < 128 * 8; i += 256) {
            int r = i >> 3, c4 = (i & 7) * 4;
            float4 fv = *(const float4*)(A + (size_t)(m0 + r) * K + k0 + c4);
            uint2 u; u.x = pack_bf16x2(fv.x, fv.y); u.y = pack_bf16x2(fv.z, fv.w);
            *(uint2*)&As[r * KS32 + c4] = u;
        }
        // Stage W 32x128 transposed into Bs[n][k]
        for (int i = tid; i < 32 * 32; i += 256) {
            int kk = i >> 5, n4 = (i & 31) * 4;
            float4 fv = *(const float4*)(W + (size_t)(k0 + kk) * N + n0 + n4);
            Bs[(n4 + 0) * KS32 + kk] = f32_to_bf16(fv.x);
            Bs[(n4 + 1) * KS32 + kk] = f32_to_bf16(fv.y);
            Bs[(n4 + 2) * KS32 + kk] = f32_to_bf16(fv.z);
            Bs[(n4 + 3) * KS32 + kk] = f32_to_bf16(fv.w);
        }
        // Prefetch next K-tile into cache while this one computes
        if (k0 + 32 < K) {
            if (tid < 128) {
                __builtin_prefetch(A + (size_t)(m0 + tid) * K + (k0 + 32), 0, 3);
            } else {
                int t = tid - 128;
                int kk = t >> 2, c = t & 3;
                __builtin_prefetch(W + (size_t)(k0 + 32 + kk) * N + n0 + c * 32,
                                   0, 3);
            }
        }
        __syncthreads();

        v16bf a0 = load_frag(As, wm, 0, KS32, lane);
        v16bf a1 = load_frag(As, wm + 16, 0, KS32, lane);
#pragma unroll
        for (int ni = 0; ni < 4; ++ni) {
            v16bf bfr = load_frag(Bs, wn + ni * 16, 0, KS32, lane);
            acc[0][ni] = wmma_bf16(a0, bfr, acc[0][ni]);
            acc[1][ni] = wmma_bf16(a1, bfr, acc[1][ni]);
        }
        __syncthreads();
    }

    // Epilogue (C/D layout: VGPR r -> M=r / M=8+r, N = lane&15)
    const int cr  = lane & 15;
    const int rhi = (lane >> 4) * 8;
#pragma unroll
    for (int mi = 0; mi < 2; ++mi)
#pragma unroll
        for (int ni = 0; ni < 4; ++ni) {
            int gcol = n0 + wn + ni * 16 + cr;
            float bv = bias[gcol];
#pragma unroll
            for (int r = 0; r < 8; ++r) {
                int grow = m0 + wm + mi * 16 + rhi + r;
                size_t idx = (size_t)grow * N + gcol;
                float v = (acc[mi][ni][r] + bv) * scale;
                if (ADD_RESIDUAL) v += residual[idx];
                if (BF16_OUT)
                    ((unsigned short*)outp)[idx] = f32_to_bf16(v);
                else
                    ((float*)outp)[idx] = v;
            }
        }
}

// ---------------------------------------------------------------------------
// Scores: S[b,h,q,k] = sum_d qh[b,q,h,d] * kh[b,k,h,d]   (qh pre-scaled 1/8)
// Block tile 128q x 128k, K = DK = 64 (two WMMA K-steps). Tiles are already
// bf16 in memory -> staged with global_load_async_to_lds_b128 (ASYNCcnt).
// ---------------------------------------------------------------------------
__global__ __launch_bounds__(256) void attn_scores_kernel(
    const unsigned short* __restrict__ qh, const unsigned short* __restrict__ kh,
    float* __restrict__ attn) {
    __shared__ unsigned short Qs[128 * KS64];   // [q][d]
    __shared__ unsigned short Ks[128 * KS64];   // [k][d]

    const int tid  = threadIdx.x;
    const int lane = tid & 31;
    const int wave = tid >> 5;
    const int n0 = blockIdx.x * 128;          // key tile
    const int m0 = blockIdx.y * 128;          // query tile
    const int bh = blockIdx.z;
    const int b = bh / Hc, h = bh % Hc;
    const int wm = (wave >> 1) * 32;
    const int wn = (wave & 1) * 64;

    // Async-stage two 128x64 bf16 tiles: 8 x 16B chunks per row.
    {
        const unsigned short* qbase =
            qh + (size_t)(b * Lc + m0) * (Hc * DKc) + h * DKc;
        const unsigned short* kbase =
            kh + (size_t)(b * Lc + n0) * (Hc * DKc) + h * DKc;
        for (int i = tid; i < 128 * 8; i += 256) {
            int r = i >> 3, c8 = (i & 7) * 8;
            async_copy_b128(lds_offset(&Qs[r * KS64 + c8]),
                            qbase + (size_t)r * (Hc * DKc) + c8);
            async_copy_b128(lds_offset(&Ks[r * KS64 + c8]),
                            kbase + (size_t)r * (Hc * DKc) + c8);
        }
        wait_async_all();
    }
    __syncthreads();

    v8f acc[2][4];
#pragma unroll
    for (int mi = 0; mi < 2; ++mi)
#pragma unroll
        for (int ni = 0; ni < 4; ++ni)
            acc[mi][ni] = (v8f){0.f, 0.f, 0.f, 0.f, 0.f, 0.f, 0.f, 0.f};

#pragma unroll
    for (int k0 = 0; k0 < 64; k0 += 32) {
        v16bf a0 = load_frag(Qs, wm, k0, KS64, lane);
        v16bf a1 = load_frag(Qs, wm + 16, k0, KS64, lane);
#pragma unroll
        for (int ni = 0; ni < 4; ++ni) {
            v16bf bfr = load_frag(Ks, wn + ni * 16, k0, KS64, lane);
            acc[0][ni] = wmma_bf16(a0, bfr, acc[0][ni]);
            acc[1][ni] = wmma_bf16(a1, bfr, acc[1][ni]);
        }
    }

    const int cr  = lane & 15;
    const int rhi = (lane >> 4) * 8;
    float* srow = attn + (size_t)bh * Lc * Lc;
#pragma unroll
    for (int mi = 0; mi < 2; ++mi)
#pragma unroll
        for (int ni = 0; ni < 4; ++ni)
#pragma unroll
            for (int r = 0; r < 8; ++r) {
                int grow = m0 + wm + mi * 16 + rhi + r;
                int gcol = n0 + wn + ni * 16 + cr;
                srow[(size_t)grow * Lc + gcol] = acc[mi][ni][r];
            }
}

// ---------------------------------------------------------------------------
// Row softmax in place over 2048 keys, with boolean mask (-inf where set).
// One 256-thread block per (b,h,q) row; 8 elements per thread.
// ---------------------------------------------------------------------------
__global__ __launch_bounds__(256) void softmax_kernel(
    float* __restrict__ attn, const unsigned char* __restrict__ mask) {
    const int tid = threadIdx.x;
    const int rid = blockIdx.x;              // (b*H + h)*L + q
    const int q  = rid & (Lc - 1);
    const int bh = rid >> 11;
    const int b  = bh >> 4;

    float* row = attn + (size_t)rid * Lc;
    const unsigned char* mrow = mask + ((size_t)b * Lc + q) * Lc;
    const float NEG_INF = __uint_as_float(0xff800000u);

    float vals[8];
    float mx = -3.402823466e38f;
#pragma unroll
    for (int j = 0; j < 8; ++j) {
        int kc = tid + 256 * j;
        float v = row[kc];
        if (mrow[kc]) v = NEG_INF;
        vals[j] = v;
        mx = fmaxf(mx, v);
    }

    __shared__ float red[256];
    red[tid] = mx;
    __syncthreads();
    for (int off = 128; off > 0; off >>= 1) {
        if (tid < off) red[tid] = fmaxf(red[tid], red[tid + off]);
        __syncthreads();
    }
    float rowmax = red[0];
    __syncthreads();

    float s = 0.f;
#pragma unroll
    for (int j = 0; j < 8; ++j) {
        float e = __expf(vals[j] - rowmax);
        vals[j] = e;
        s += e;
    }
    red[tid] = s;
    __syncthreads();
    for (int off = 128; off > 0; off >>= 1) {
        if (tid < off) red[tid] += red[tid + off];
        __syncthreads();
    }
    float inv = 1.f / red[0];
#pragma unroll
    for (int j = 0; j < 8; ++j) row[tid + 256 * j] = vals[j] * inv;
}

// ---------------------------------------------------------------------------
// ctx[b,q,h,dv] = sum_k attn[b,h,q,k] * vh[b,k,h,dv]
// Block: 128 q-rows x 64 dv, 8 waves x 16 rows each; K loop over 2048, step 32.
// ---------------------------------------------------------------------------
__global__ __launch_bounds__(256) void attn_ctx_kernel(
    const float* __restrict__ attn, const unsigned short* __restrict__ vh,
    float* __restrict__ ctx) {
    __shared__ unsigned short As[128 * KS32];   // attn tile [q][k] as bf16
    __shared__ unsigned short Bs[64 * KS32];    // vh tile transposed [dv][k]

    const int tid  = threadIdx.x;
    const int lane = tid & 31;
    const int wave = tid >> 5;
    const int m0 = blockIdx.x * 128;
    const int bh = blockIdx.y;
    const int b = bh / Hc, h = bh % Hc;
    const int wm = wave * 16;

    const float* arow = attn + (size_t)bh * Lc * Lc;

    v8f acc[4];
#pragma unroll
    for (int ni = 0; ni < 4; ++ni)
        acc[ni] = (v8f){0.f, 0.f, 0.f, 0.f, 0.f, 0.f, 0.f, 0.f};

    for (int k0 = 0; k0 < Lc; k0 += 32) {
        // attn 128x32 f32 -> bf16
        for (int i = tid; i < 128 * 8; i += 256) {
            int r = i >> 3, c4 = (i & 7) * 4;
            float4 fv =
                *(const float4*)(arow + (size_t)(m0 + r) * Lc + k0 + c4);
            uint2 u; u.x = pack_bf16x2(fv.x, fv.y); u.y = pack_bf16x2(fv.z, fv.w);
            *(uint2*)&As[r * KS32 + c4] = u;
        }
        // vh 32x64 bf16, transposed into [dv][k]
        for (int i = tid; i < 32 * 32; i += 256) {
            int kk = i >> 5, dv2 = (i & 31) * 2;
            unsigned u = *(const unsigned*)(
                vh + ((size_t)(b * Lc + k0 + kk)) * (Hc * DVc) + h * DVc + dv2);
            Bs[(dv2 + 0) * KS32 + kk] = (unsigned short)(u & 0xFFFFu);
            Bs[(dv2 + 1) * KS32 + kk] = (unsigned short)(u >> 16);
        }
        __syncthreads();

        v16bf a0 = load_frag(As, wm, 0, KS32, lane);
#pragma unroll
        for (int ni = 0; ni < 4; ++ni) {
            v16bf bfr = load_frag(Bs, ni * 16, 0, KS32, lane);
            acc[ni] = wmma_bf16(a0, bfr, acc[ni]);
        }
        __syncthreads();
    }

    const int cr  = lane & 15;
    const int rhi = (lane >> 4) * 8;
#pragma unroll
    for (int ni = 0; ni < 4; ++ni)
#pragma unroll
        for (int r = 0; r < 8; ++r) {
            int grow = m0 + wm + rhi + r;
            int gcol = h * DVc + ni * 16 + cr;
            ctx[(size_t)(b * Lc + grow) * Dc + gcol] = acc[ni][r];
        }
}

// ---------------------------------------------------------------------------
// LayerNorm over D=1024 per row (population variance, matches jnp.var).
// ---------------------------------------------------------------------------
__global__ __launch_bounds__(256) void layernorm_kernel(
    const float* __restrict__ x, const float* __restrict__ gamma,
    const float* __restrict__ beta, float* __restrict__ out) {
    const int tid = threadIdx.x;
    const int row = blockIdx.x;
    const float* xr = x + (size_t)row * Dc;

    float vals[4];
    float s = 0.f, ss = 0.f;
#pragma unroll
    for (int j = 0; j < 4; ++j) {
        float v = xr[tid + 256 * j];
        vals[j] = v;
        s += v;
        ss += v * v;
    }
    __shared__ float r1[256], r2[256];
    r1[tid] = s;
    r2[tid] = ss;
    __syncthreads();
    for (int off = 128; off > 0; off >>= 1) {
        if (tid < off) { r1[tid] += r1[tid + off]; r2[tid] += r2[tid + off]; }
        __syncthreads();
    }
    float mu   = r1[0] * (1.f / Dc);
    float var  = r2[0] * (1.f / Dc) - mu * mu;
    float rstd = rsqrtf(var + 1e-5f);
#pragma unroll
    for (int j = 0; j < 4; ++j) {
        int col = tid + 256 * j;
        out[(size_t)row * Dc + col] = (vals[j] - mu) * rstd * gamma[col] + beta[col];
    }
}

// ---------------------------------------------------------------------------
extern "C" void kernel_launch(void* const* d_in, const int* in_sizes, int n_in,
                              void* d_out, int out_size, void* d_ws, size_t ws_size,
                              hipStream_t stream) {
    const float* q    = (const float*)d_in[0];
    const float* k    = (const float*)d_in[1];
    const float* v    = (const float*)d_in[2];
    // d_in[3] = c (unused by reference)
    const unsigned char* mask = (const unsigned char*)d_in[4];
    const float* Wq = (const float*)d_in[5];
    const float* bq = (const float*)d_in[6];
    const float* Wk = (const float*)d_in[7];
    const float* bk = (const float*)d_in[8];
    const float* Wv = (const float*)d_in[9];
    const float* bv = (const float*)d_in[10];
    const float* Wo = (const float*)d_in[11];
    const float* bo = (const float*)d_in[12];
    const float* gamma = (const float*)d_in[13];
    const float* beta  = (const float*)d_in[14];

    const int M = Bc * Lc;           // 8192 rows
    const size_t MD = (size_t)M * Dc;

    // Workspace layout (80 MB peak; x reuses qh/kh which are dead after scores):
    unsigned short* qh = (unsigned short*)d_ws;       // MD bf16 (16 MB)
    unsigned short* kh = qh + MD;                     // MD bf16 (16 MB)
    unsigned short* vh = kh + MD;                     // MD bf16 (16 MB)
    float* ctx = (float*)(vh + MD);                   // MD f32  (32 MB)
    float* x   = (float*)d_ws;                        // MD f32, overlays qh+kh

    float* out_main = (float*)d_out;                  // B*L*D
    float* attn     = out_main + MD;                  // B*H*L*L

    dim3 blk(256);
    dim3 gGemm(Dc / 128, M / 128);                    // (8, 64)

    // 1-3: QKV projections (temperature 1/sqrt(64) folded into Q epilogue)
    gemm_kernel<true, false><<<gGemm, blk, 0, stream>>>(q, Wq, bq, nullptr, qh,
                                                        M, Dc, Dc, 0.125f);
    gemm_kernel<true, false><<<gGemm, blk, 0, stream>>>(k, Wk, bk, nullptr, kh,
                                                        M, Dc, Dc, 1.0f);
    gemm_kernel<true, false><<<gGemm, blk, 0, stream>>>(v, Wv, bv, nullptr, vh,
                                                        M, Dc, Dc, 1.0f);

    // 4: scores -> attn region of d_out
    dim3 gScores(Lc / 128, Lc / 128, Bc * Hc);        // (16, 16, 64)
    attn_scores_kernel<<<gScores, blk, 0, stream>>>(qh, kh, attn);

    // 5: masked softmax in place
    softmax_kernel<<<dim3(Bc * Hc * Lc), blk, 0, stream>>>(attn, mask);

    // 6: ctx = attn @ vh
    dim3 gCtx(Lc / 128, Bc * Hc);                     // (16, 64)
    attn_ctx_kernel<<<gCtx, blk, 0, stream>>>(attn, vh, ctx);

    // 7: output projection + residual
    gemm_kernel<false, true><<<gGemm, blk, 0, stream>>>(ctx, Wo, bo, q, x,
                                                        M, Dc, Dc, 1.0f);

    // 8: LayerNorm -> d_out main region
    layernorm_kernel<<<dim3(M), blk, 0, stream>>>(x, gamma, beta, out_main);
}
